// CompressedSensingConvolutional_33131377722035
// MI455X (gfx1250) — matrix operand order repaired and checked by reading, not compiled, not run on hardware
//
#include <hip/hip_runtime.h>
#include <hip/hip_bf16.h>
#include <math.h>

typedef __attribute__((ext_vector_type(16))) _Float16 v16h;
typedef __attribute__((ext_vector_type(8)))  _Float16 v8h;
typedef __attribute__((ext_vector_type(8)))  float    v8f;

#define HW      72
#define LOW     9
#define NPIX    (HW*HW)        // 5184
#define NELEM   (NPIX*3)       // 15552
#define NITERS  200
#define PADW    104            // 72 + 2*16 padding for the 40-wide stride-8 conv
#define YH_HALVES (PADW*PADW*3 + 64)  // 32512 (incl. overrun slack for padded-K reads)
#define KPAD    5120           // 40 kernel rows * 128 (row of 120 padded to 128)
#define NCHUNK  (KPAD/32)      // 160 K-chunks of 32 for 16x16x32 WMMA
#define WBT_HALVES (4*KPAD)    // 4 N-planes (oc0..2 + zero plane)

// float region: yt,yl,cf (3*15552) + sred(256) + sw(8*256) + wct(240) = 49200 floats
#define F_REGION 49200
#define SMEM_BYTES (F_REGION*4 + WBT_HALVES*2 + YH_HALVES*2)   // 302784 B < 320KB

__launch_bounds__(256)
__global__ void fista_cs_kernel(const float* __restrict__ x,
                                const float* __restrict__ lam,
                                const float* __restrict__ w_conv,
                                const float* __restrict__ b_conv,
                                const float* __restrict__ w_ct,
                                const float* __restrict__ b_ct,
                                float* __restrict__ out)
{
    extern __shared__ char smem[];
    float*    yt   = (float*)smem;          // y_tmp   (fp32)
    float*    yl   = yt + NELEM;            // y_last  (fp32)
    float*    cf   = yl + NELEM;            // c field (fp32)
    float*    sred = cf + NELEM;            // reduced convS output (243 used)
    float*    sw   = sred + 256;            // per-wave partials: 8 x 256
    float*    wct  = sw + 8*256;            // conv-transpose weights (225 used)
    _Float16* wbt  = (_Float16*)(wct + 240);// B-matrix planes [4][KPAD], f16
    _Float16* yh   = wbt + WBT_HALVES;      // padded f16 y_tmp [104][104][3]

    const int tid  = threadIdx.x;
    const int b    = blockIdx.x;
    const int lane = tid & 31;
    const int wave = tid >> 5;
    const int hi   = lane >> 4;      // which K-half of the fragment this lane holds
    const int mrow = lane & 15;      // A row / B,D column index

    const float bc0 = b_conv[0], bc1 = b_conv[1], bc2 = b_conv[2];
    const float bt0 = b_ct[0],   bt1 = b_ct[1],   bt2 = b_ct[2];
    const float lamb = lam[b];
    const float* xb  = x + b*(LOW*LOW*3);

    // ---- one-time setup ------------------------------------------------
    for (int i = tid; i < 225; i += 256) wct[i] = w_ct[i];
    for (int i = tid; i < YH_HALVES; i += 256) yh[i] = (_Float16)0.0f;
    for (int i = tid; i < NELEM; i += 256) { yt[i] = 0.f; yl[i] = 0.f; }
    // B planes: wbt[n][k'] ; k' = ky*128 + (kx*3+ic), pad region & n==3 plane are zero
    for (int idx = tid; idx < WBT_HALVES; idx += 256) {
        int n  = idx / KPAD;
        int kp = idx - n*KPAD;
        int ky = kp >> 7, rem = kp & 127;
        float v = 0.f;
        if (n < 3 && rem < 120) v = w_conv[(ky*120 + rem)*3 + n];
        wbt[idx] = (_Float16)v;
    }
    __syncthreads();
    // c = convT(x - b_conv) + b_ct  (conv_transpose SAME,s=8,k=5 -> ky=4-d anchor)
    for (int pix = tid; pix < NPIX; pix += 256) {
        int Y = pix / HW, X = pix - (pix/HW)*HW;
        int i = Y >> 3, d = Y & 7, j = X >> 3, e = X & 7;
        float m0 = 0.f, m1 = 0.f, m2 = 0.f;
        if (d < 5 && e < 5) {
            int wb = ((4-d)*5 + (4-e))*9;
            const float* xp = xb + (i*LOW + j)*3;
            float r0 = xp[0]-bc0, r1 = xp[1]-bc1, r2 = xp[2]-bc2;
            m0 = r0*wct[wb+0] + r1*wct[wb+3] + r2*wct[wb+6];
            m1 = r0*wct[wb+1] + r1*wct[wb+4] + r2*wct[wb+7];
            m2 = r0*wct[wb+2] + r1*wct[wb+5] + r2*wct[wb+8];
        }
        cf[pix*3+0] = m0 + bt0;
        cf[pix*3+1] = m1 + bt1;
        cf[pix*3+2] = m2 + bt2;
    }
    // per-lane patch base addresses (constant across all iterations)
    int lb[6];
#pragma unroll
    for (int t = 0; t < 6; ++t) {
        int pos = t*16 + mrow; if (pos > 80) pos = 80;   // clamp; masked on extract
        int i = pos / 9, j = pos - (pos/9)*9;
        lb[t] = 24 * (i*PADW + j);                       // (8i*104 + 8j)*3
    }
    const _Float16* bbase = wbt + ((mrow < 3) ? mrow : 3) * KPAD;

    // ---- 200 FISTA iterations, fully LDS-resident ----------------------
    float tmom = 1.0f;
    for (int it = 0; it < NITERS; ++it) {
        __syncthreads();                 // yh from previous update visible
        v8f acc[6] = {};
        for (int c = wave; c < NCHUNK; c += 8) {
            // B fragment: K = hi*16 + e, contiguous halves in lane's plane
            int kb = c*32 + hi*16;
            v8h b0 = *(const v8h*)(bbase + kb);
            v8h b1 = *(const v8h*)(bbase + kb + 8);
            v16h Bf = __builtin_shufflevector(b0, b1,
                0,1,2,3,4,5,6,7,8,9,10,11,12,13,14,15);
            // A fragment runs: K = hi*16 + blk*8 + w ; row-padded so each run
            // of 8 halves is contiguous & 16B aligned -> ds_load_b128
            int r0 = c*32 + hi*16;
            int r1 = r0 + 8;
            int off0 = (r0 >> 7)*312 + (r0 & 127);
            int off1 = (r1 >> 7)*312 + (r1 & 127);
#pragma unroll
            for (int t = 0; t < 6; ++t) {
                v8h a0 = *(const v8h*)(yh + lb[t] + off0);
                v8h a1 = *(const v8h*)(yh + lb[t] + off1);
                v16h Af = __builtin_shufflevector(a0, a1,
                    0,1,2,3,4,5,6,7,8,9,10,11,12,13,14,15);
                acc[t] = __builtin_amdgcn_wmma_f32_16x16x32_f16(
                    false, Af, false, Bf, (short)0, acc[t], false, false);
            }
        }
        // write per-wave partials (D: M = hi*8 + r, N = mrow)
        float* swp = sw + wave*256;
#pragma unroll
        for (int t = 0; t < 6; ++t) {
#pragma unroll
            for (int r = 0; r < 8; ++r) {
                int pos = t*16 + hi*8 + r;
                if (mrow < 3 && pos < 81) swp[pos*3 + mrow] = acc[t][r];
            }
        }
        __syncthreads();
        for (int p = tid; p < 243; p += 256) {
            float v = sw[p]        + sw[256+p]  + sw[512+p]  + sw[768+p]
                    + sw[1024+p]   + sw[1280+p] + sw[1536+p] + sw[1792+p];
            sred[p] = v;
        }
        __syncthreads();
        // elementwise: w = y + convT(s) - c ; softthresh ; momentum
        float tn   = 0.5f*(1.0f + sqrtf(1.0f + 4.0f*tmom*tmom));
        float coef = (tmom - 1.0f)/tn;
        tmom = tn;
        for (int pix = tid; pix < NPIX; pix += 256) {
            int Y = pix / HW, X = pix - (pix/HW)*HW;
            int i = Y >> 3, d = Y & 7, j = X >> 3, e = X & 7;
            float m0 = 0.f, m1 = 0.f, m2 = 0.f;
            if (d < 5 && e < 5) {
                int wb = ((4-d)*5 + (4-e))*9;
                const float* sp = sred + (i*9 + j)*3;
                float s0 = sp[0], s1 = sp[1], s2 = sp[2];
                m0 = s0*wct[wb+0] + s1*wct[wb+3] + s2*wct[wb+6];
                m1 = s0*wct[wb+1] + s1*wct[wb+4] + s2*wct[wb+7];
                m2 = s0*wct[wb+2] + s1*wct[wb+5] + s2*wct[wb+8];
            }
            int idx = pix*3;
            int ph  = ((Y+16)*PADW + (X+16))*3;
#pragma unroll
            for (int oc = 0; oc < 3; ++oc) {
                float mm = (oc==0) ? m0 : ((oc==1) ? m1 : m2);
                float wv = yt[idx+oc] + mm - cf[idx+oc];
                float aw = fabsf(wv) - lamb;
                float yn = (aw > 0.f) ? copysignf(aw, wv) : 0.f;
                float ytn = yn + coef*(yn - yl[idx+oc]);
                yl[idx+oc] = yn;
                yt[idx+oc] = ytn;
                yh[ph+oc]  = (_Float16)ytn;
            }
        }
    }
    __syncthreads();
    float* ob = out + b*NELEM;
    for (int i = tid; i < NELEM; i += 256) ob[i] = yl[i];
}

extern "C" void kernel_launch(void* const* d_in, const int* in_sizes, int n_in,
                              void* d_out, int out_size, void* d_ws, size_t ws_size,
                              hipStream_t stream) {
    const float* x      = (const float*)d_in[0];
    const float* lamv   = (const float*)d_in[1];
    const float* w_conv = (const float*)d_in[2];
    const float* b_conv = (const float*)d_in[3];
    const float* w_ct   = (const float*)d_in[4];
    const float* b_ct   = (const float*)d_in[5];
    float* out = (float*)d_out;

    (void)hipFuncSetAttribute((const void*)fista_cs_kernel,
                              hipFuncAttributeMaxDynamicSharedMemorySize,
                              SMEM_BYTES);
    fista_cs_kernel<<<64, 256, SMEM_BYTES, stream>>>(
        x, lamv, w_conv, b_conv, w_ct, b_ct, out);
}
